// RADFE_28260884807861
// MI455X (gfx1250) — matrix operand order repaired and checked by reading, not compiled, not run on hardware
//
#include <hip/hip_runtime.h>
#include <hip/hip_bf16.h>
#include <math.h>

typedef __attribute__((ext_vector_type(16))) _Float16 v16h;
typedef __attribute__((ext_vector_type(8)))  _Float16 v8h;
typedef __attribute__((ext_vector_type(8)))  float    v8f;
typedef __attribute__((ext_vector_type(4)))  unsigned int v4u;
typedef __attribute__((ext_vector_type(4)))  int v4i;
typedef __attribute__((ext_vector_type(8)))  int v8i;

__device__ __forceinline__ float sigf(float x)  { return 1.f / (1.f + __expf(-x)); }
__device__ __forceinline__ float siluf(float x) { return x * sigf(x); }
__device__ __forceinline__ float softplusf(float x) { return (x > 20.f) ? x : log1pf(__expf(x)); }

// ---------------------------------------------------------------------------
// Pack a row-major weight W[Nsrc][K] (f32) into WMMA B-fragment order (f16).
// Fragment layout per CDNA5 ISA (16-bit B, 32x16):
//   lane<16 : half h -> K = kc*32 + (h<8 ? h : h+8),     N = ntile*16 + lane
//   lane>=16: half h -> K = kc*32 + 8 + (h<8 ? h : h+8), N = ntile*16 + (lane-16)
// ---------------------------------------------------------------------------
__global__ void k_packB(const float* __restrict__ W, _Float16* __restrict__ out,
                        int Nsrc, int K, int KC, int total) {
  int idx = blockIdx.x * blockDim.x + threadIdx.x;
  if (idx >= total) return;
  int h     = idx & 15;
  int lane  = (idx >> 4) & 31;
  int rem   = idx >> 9;        // ntile*KC + kc
  int kc    = rem % KC;
  int ntile = rem / KC;
  int k = kc * 32 + ((lane < 16) ? 0 : 8) + ((h < 8) ? h : h + 8);
  int n = ntile * 16 + (lane & 15);
  float v = (n < Nsrc) ? W[(size_t)n * K + k] : 0.f;
  out[idx] = (_Float16)v;
}

// ---------------------------------------------------------------------------
// Generic WMMA GEMM: C[T][N] = act(A[T][K] * B + bias), A row-major f16,
// B pre-packed fragments, f32 accumulation via v_wmma_f32_16x16x32_f16.
// One wave -> one 16x16 output tile. Block = 8 waves.
// ---------------------------------------------------------------------------
__global__ void __launch_bounds__(256) k_gemm(
    const _Float16* __restrict__ A, const _Float16* __restrict__ Bp,
    float* __restrict__ Cf, _Float16* __restrict__ Ch,
    const float* __restrict__ bias, int T, int K, int N, int act) {
  int lane  = threadIdx.x & 31;
  int wave  = threadIdx.x >> 5;
  int m0    = (blockIdx.x * 8 + wave) * 16;
  int ntile = blockIdx.y;
  int KC    = K >> 5;
  if (m0 >= T) return;

  v8f acc = {};
  // A fragment: lane<16 holds K halves {0..7,16..23}, lane>=16 {8..15,24..31}
  const _Float16* pa0 = A + (size_t)(m0 + (lane & 15)) * K + ((lane & 16) ? 8 : 0);
  const _Float16* pb0 = Bp + ((size_t)ntile * KC * 32 + lane) * 16;

  for (int kc = 0; kc < KC; ++kc) {
    const _Float16* pa = pa0 + kc * 32;
    v8h lo = *(const v8h*)pa;
    v8h hi = *(const v8h*)(pa + 16);
    v16h a;
#pragma unroll
    for (int i = 0; i < 8; ++i) { a[i] = lo[i]; a[i + 8] = hi[i]; }
    v16h b = *(const v16h*)(pb0 + (size_t)kc * 32 * 16);
    acc = __builtin_amdgcn_wmma_f32_16x16x32_f16(false, a, false, b,
                                                 (short)0, acc, false, false);
  }

  int n  = ntile * 16 + (lane & 15);
  int mb = m0 + ((lane & 16) ? 8 : 0);
  float bv = bias ? bias[n] : 0.f;
#pragma unroll
  for (int r = 0; r < 8; ++r) {
    float v = acc[r] + bv;
    if (act == 1) v = siluf(v);
    size_t o = (size_t)(mb + r) * N + n;
    if (Cf) Cf[o] = v;
    if (Ch) Ch[o] = (_Float16)v;
  }
}

// ---------------------------------------------------------------------------
// Fast-path token prep: LN -> fast_lin+silu -> LN -> +fast_pe, transpose
// (1,F,S,C) -> tokens [s*512+f][32], f16 output for WMMA.
// ---------------------------------------------------------------------------
__global__ void __launch_bounds__(256) k_fast_tokens(
    const float* __restrict__ x, const float* __restrict__ g0, const float* __restrict__ b0,
    const float* __restrict__ lw, const float* __restrict__ lb,
    const float* __restrict__ g1, const float* __restrict__ b1,
    const float* __restrict__ pe, _Float16* __restrict__ xt) {
  __shared__ float sW[1024], sg0[32], sb0[32], sg1[32], sb1[32], slb[32];
  for (int i = threadIdx.x; i < 1024; i += blockDim.x) sW[i] = lw[i];
  if (threadIdx.x < 32) {
    int c = threadIdx.x;
    sg0[c] = g0[c]; sb0[c] = b0[c]; sg1[c] = g1[c]; sb1[c] = b1[c]; slb[c] = lb[c];
  }
  __syncthreads();
  int token = blockIdx.x * blockDim.x + threadIdx.x;
  if (token >= 131072) return;
  int s = token >> 9, f = token & 511;
  const float* px = x + ((size_t)f * 256 + s) * 32;

  float v[32];
#pragma unroll
  for (int c = 0; c < 32; ++c) v[c] = px[c];
  float mu = 0.f;
#pragma unroll
  for (int c = 0; c < 32; ++c) mu += v[c];
  mu *= (1.f / 32.f);
  float var = 0.f;
#pragma unroll
  for (int c = 0; c < 32; ++c) { float d = v[c] - mu; var += d * d; }
  float inv = rsqrtf(var * (1.f / 32.f) + 1e-5f);
#pragma unroll
  for (int c = 0; c < 32; ++c) v[c] = (v[c] - mu) * inv * sg0[c] + sb0[c];

  float o[32];
#pragma unroll
  for (int oc = 0; oc < 32; ++oc) {
    float a = slb[oc];
    for (int ic = 0; ic < 32; ++ic) a += sW[oc * 32 + ic] * v[ic];
    o[oc] = siluf(a);
  }
  mu = 0.f;
#pragma unroll
  for (int c = 0; c < 32; ++c) mu += o[c];
  mu *= (1.f / 32.f);
  var = 0.f;
#pragma unroll
  for (int c = 0; c < 32; ++c) { float d = o[c] - mu; var += d * d; }
  inv = rsqrtf(var * (1.f / 32.f) + 1e-5f);
  _Float16* po = xt + (size_t)token * 32;
#pragma unroll
  for (int c = 0; c < 32; ++c)
    po[c] = (_Float16)((o[c] - mu) * inv * sg1[c] + sb1[c] + pe[f * 32 + c]);
}

// depthwise causal conv over fast dim; input = xz cols 0..63, output f16 (after silu)
__global__ void k_conv1d(const _Float16* __restrict__ xz, _Float16* __restrict__ xc,
                         const float* __restrict__ cw, const float* __restrict__ cb,
                         int T, int L) {
  int i = blockIdx.x * blockDim.x + threadIdx.x;
  if (i >= T * 64) return;
  int tok = i >> 6, d = i & 63;
  int f = tok % L;
  float acc = cb[d];
#pragma unroll
  for (int j = 0; j < 4; ++j) {
    int t = f + j - 3;
    if (t >= 0) acc += cw[d * 4 + j] * (float)xz[(size_t)(tok + j - 3) * 128 + d];
  }
  xc[i] = (_Float16)siluf(acc);
}

// ---------------------------------------------------------------------------
// Fused selective scan: dt-projection (softplus) + scan + (y+u*D)*silu(z).
// Block = one sequence, 64 threads = channels; B/C/dt_raw staged in LDS.
// Next rows of x_dbl prefetched via global_prefetch + TDM tensor_load_to_lds.
// ---------------------------------------------------------------------------
__global__ void __launch_bounds__(64) k_scan(
    const float* __restrict__ xd, const _Float16* __restrict__ u16,
    const _Float16* __restrict__ xz,
    const float* __restrict__ dtw, const float* __restrict__ dtb,
    const float* __restrict__ A_log, const float* __restrict__ Dp,
    _Float16* __restrict__ yg, int L) {
  __shared__ float shB[32];
  __shared__ float shC[32];
  __shared__ float shDt[2];
  __shared__ float tdm_buf[16 * 66];   // TDM staging/prefetch target

  int d = threadIdx.x;
  int seq = blockIdx.x;
  float w0 = dtw[d * 2 + 0], w1 = dtw[d * 2 + 1], bb = dtb[d], Dd = Dp[d];
  float Ar[32], hs[32];
#pragma unroll
  for (int s = 0; s < 32; ++s) { Ar[s] = -__expf(A_log[d * 32 + s]); hs[s] = 0.f; }

  for (int f = 0; f < L; ++f) {
    size_t tok = (size_t)seq * L + f;
    if (d < 32) shB[d] = xd[tok * 80 + 2 + d];
    else        shC[d - 32] = xd[tok * 80 + 34 + (d - 32)];
    if (d < 2)  shDt[d] = xd[tok * 80 + d];

    // prefetch the x_dbl row 8 steps ahead into cache
    if (d == 0 && f + 8 < L) __builtin_prefetch(xd + (tok + 8) * 80, 0, 1);

#if defined(__has_builtin)
#if __has_builtin(__builtin_amdgcn_tensor_load_to_lds) && __has_builtin(__builtin_amdgcn_s_wait_tensorcnt)
    // TDM: async-stage the next 16 x_dbl rows (66 valid floats each, stride 80)
    // into LDS. D# per CDNA5 ISA: count=1, type=2 (image),
    // data_size=4B, tile 66x16, tensor_dim0=66, dim0_stride=80.
    if (threadIdx.x == 0 && (f & 15) == 0 && f + 32 <= L) {
      unsigned long long ga =
          (unsigned long long)(uintptr_t)(xd + (tok + 16) * 80);
      unsigned int ldsoff = (unsigned int)(uintptr_t)(void*)tdm_buf;
      v4u g0;
      g0[0] = 1u;                                   // count=1 valid descriptor
      g0[1] = ldsoff;                               // lds_addr
      g0[2] = (unsigned int)ga;                     // global_addr[31:0]
      g0[3] = (unsigned int)(ga >> 32) | (2u << 30);// global_addr[56:32] | type=2
      v8i g1;
      g1[0] = 0x20000;       // workgroup_mask=0, data_size=2 (4 bytes)
      g1[1] = (66 << 16);    // tensor_dim0[15:0]=66
      g1[2] = (16 << 16);    // tensor_dim0 hi=0, tensor_dim1[15:0]=16
      g1[3] = (66 << 16);    // tensor_dim1 hi=0, tile_dim0=66
      g1[4] = 16;            // tile_dim1=16, tile_dim2=0
      g1[5] = 80;            // tensor_dim0_stride=80
      g1[6] = 0;
      g1[7] = 0;
      v4i gz = {0, 0, 0, 0};
      __builtin_amdgcn_s_wait_tensorcnt(2);         // throttle outstanding TDM
#if __clang_major__ >= 23
      v8i gz8 = {0, 0, 0, 0, 0, 0, 0, 0};
      __builtin_amdgcn_tensor_load_to_lds(g0, g1, gz, gz, gz8, 0);
#else
      __builtin_amdgcn_tensor_load_to_lds(g0, g1, gz, gz, 0);
#endif
    }
#endif
#endif
    __syncthreads();

    float dt = softplusf(w0 * shDt[0] + w1 * shDt[1] + bb);
    float u  = (float)u16[tok * 64 + d];
    float z  = (float)xz[tok * 128 + 64 + d];
    float du = dt * u;
    float yv = 0.f;
#pragma unroll
    for (int s = 0; s < 32; ++s) {
      float dA = __expf(dt * Ar[s]);
      hs[s] = dA * hs[s] + du * shB[s];
      yv += hs[s] * shC[s];
    }
    yg[tok * 64 + d] = (_Float16)((yv + u * Dd) * siluf(z));
    __syncthreads();
  }
}

// chirp = mean over fast axis of fast-mamba output
__global__ void k_chirp(const _Float16* __restrict__ oh, float* __restrict__ chirp) {
  int idx = blockIdx.x * blockDim.x + threadIdx.x;
  if (idx >= 256 * 32) return;
  int s = idx >> 5, c = idx & 31;
  float acc = 0.f;
  for (int f = 0; f < 512; ++f) acc += (float)oh[((size_t)(s * 512 + f)) * 32 + c];
  chirp[idx] = acc * (1.f / 512.f);
}

__global__ void k_slowtok(const float* __restrict__ chirp, const float* __restrict__ pe,
                          _Float16* __restrict__ xt) {
  int i = blockIdx.x * blockDim.x + threadIdx.x;
  if (i >= 8192) return;
  xt[i] = (_Float16)(chirp[i] + pe[i]);
}

// feat = mean_t(concat(x_enc, chirp)) @ {proj_w, projD_w}.T + bias  -> feat[2048]
__global__ void __launch_bounds__(256) k_feat(
    const float* __restrict__ xenc, const float* __restrict__ chirp,
    const float* __restrict__ pw, const float* __restrict__ pb,
    const float* __restrict__ pwD, const float* __restrict__ pbD,
    float* __restrict__ feat) {
  __shared__ float m[64];
  int t = threadIdx.x;
  if (t < 64) {
    const float* src = (t < 32) ? xenc : chirp;
    int c = t & 31;
    float s = 0.f;
    for (int i = 0; i < 256; ++i) s += src[i * 32 + c];
    m[t] = s * (1.f / 256.f);
  }
  __syncthreads();
  for (int i = t; i < 2048; i += 256) {
    const float* W; const float* B; int j;
    if (i < 1024) { W = pw;  B = pb;  j = i; }
    else          { W = pwD; B = pbD; j = i - 1024; }
    float a = B[j];
    for (int k = 0; k < 64; ++k) a += m[k] * W[j * 64 + k];
    feat[i] = a;
  }
}

// generic 3x3 SAME conv, optional inference-BN and activation (0 none,1 silu,2 relu,3 sigmoid)
__global__ void k_conv3(const float* __restrict__ in, float* __restrict__ out,
                        const float* __restrict__ w, const float* __restrict__ b,
                        const float* __restrict__ bng, const float* __restrict__ bnb,
                        int Cin, int Cout, int H, int W, int act) {
  int idx = blockIdx.x * blockDim.x + threadIdx.x;
  int HW = H * W;
  if (idx >= Cout * HW) return;
  int co = idx / HW, r = idx - co * HW, y = r / W, x = r - y * W;
  float acc = b[co];
  for (int ci = 0; ci < Cin; ++ci) {
    const float* ip = in + (size_t)ci * HW;
    const float* wp = w + ((size_t)co * Cin + ci) * 9;
#pragma unroll
    for (int ky = 0; ky < 3; ++ky) {
      int iy = y + ky - 1;
      if (iy < 0 || iy >= H) continue;
#pragma unroll
      for (int kx = 0; kx < 3; ++kx) {
        int ix = x + kx - 1;
        if (ix < 0 || ix >= W) continue;
        acc += wp[ky * 3 + kx] * ip[iy * W + ix];
      }
    }
  }
  if (bng) acc = acc * (bng[co] * rsqrtf(1.f + 1e-5f)) + bnb[co];
  if (act == 1) acc = siluf(acc);
  else if (act == 2) acc = fmaxf(acc, 0.f);
  else if (act == 3) acc = sigf(acc);
  out[idx] = acc;
}

// bilinear resize (half-pixel centers, edge clamp)
__global__ void k_up(const float* __restrict__ in, float* __restrict__ out,
                     int C, int Hi, int Wi, int Ho, int Wo) {
  int idx = blockIdx.x * blockDim.x + threadIdx.x;
  if (idx >= C * Ho * Wo) return;
  int c = idx / (Ho * Wo), r = idx - c * (Ho * Wo), y = r / Wo, x = r - y * Wo;
  float fy = ((y + 0.5f) * Hi) / Ho - 0.5f;
  float fx = ((x + 0.5f) * Wi) / Wo - 0.5f;
  fy = fminf(fmaxf(fy, 0.f), (float)(Hi - 1));
  fx = fminf(fmaxf(fx, 0.f), (float)(Wi - 1));
  int y0 = (int)fy, x0 = (int)fx;
  int y1 = (y0 + 1 < Hi) ? y0 + 1 : Hi - 1;
  int x1 = (x0 + 1 < Wi) ? x0 + 1 : Wi - 1;
  float wy = fy - y0, wx = fx - x0;
  const float* ip = in + (size_t)c * Hi * Wi;
  out[idx] = (1.f - wy) * ((1.f - wx) * ip[y0 * Wi + x0] + wx * ip[y0 * Wi + x1]) +
             wy * ((1.f - wx) * ip[y1 * Wi + x0] + wx * ip[y1 * Wi + x1]);
}

// ---------------------------------------------------------------------------
extern "C" void kernel_launch(void* const* d_in, const int* in_sizes, int n_in,
                              void* d_out, int out_size, void* d_ws, size_t ws_size,
                              hipStream_t stream) {
  (void)in_sizes; (void)n_in; (void)out_size; (void)ws_size;

  // ---- inputs (setup_inputs insertion order) ----
  const float* X          = (const float*)d_in[0];
  const float* in_g       = (const float*)d_in[1];
  const float* in_b       = (const float*)d_in[2];
  const float* fast_lin_w = (const float*)d_in[3];
  const float* fast_lin_b = (const float*)d_in[4];
  const float* fast_ln_g  = (const float*)d_in[5];
  const float* fast_ln_b  = (const float*)d_in[6];
  const float* fast_pe    = (const float*)d_in[7];
  const float* slow_pe    = (const float*)d_in[8];
  const float* f_inw  = (const float*)d_in[9];
  const float* f_cw   = (const float*)d_in[10];
  const float* f_cb   = (const float*)d_in[11];
  const float* f_xpw  = (const float*)d_in[12];
  const float* f_dtw  = (const float*)d_in[13];
  const float* f_dtb  = (const float*)d_in[14];
  const float* f_Alog = (const float*)d_in[15];
  const float* f_D    = (const float*)d_in[16];
  const float* f_opw  = (const float*)d_in[17];
  const float* f_bkw  = (const float*)d_in[18];
  const float* f_bkb  = (const float*)d_in[19];
  const float* s_inw  = (const float*)d_in[20];
  const float* s_cw   = (const float*)d_in[21];
  const float* s_cb   = (const float*)d_in[22];
  const float* s_xpw  = (const float*)d_in[23];
  const float* s_dtw  = (const float*)d_in[24];
  const float* s_dtb  = (const float*)d_in[25];
  const float* s_Alog = (const float*)d_in[26];
  const float* s_D    = (const float*)d_in[27];
  const float* s_opw  = (const float*)d_in[28];
  const float* s_bkw  = (const float*)d_in[29];
  const float* s_bkb  = (const float*)d_in[30];
  const float* proj_w  = (const float*)d_in[31];
  const float* proj_b  = (const float*)d_in[32];
  const float* c0_w  = (const float*)d_in[33];  const float* c0_b  = (const float*)d_in[34];
  const float* bn0_g = (const float*)d_in[35];  const float* bn0_b = (const float*)d_in[36];
  const float* c11_w = (const float*)d_in[37];  const float* c11_b = (const float*)d_in[38];
  const float* bn11_g = (const float*)d_in[39]; const float* bn11_b = (const float*)d_in[40];
  const float* c12_w = (const float*)d_in[41];  const float* c12_b = (const float*)d_in[42];
  const float* bn12_g = (const float*)d_in[43]; const float* bn12_b = (const float*)d_in[44];
  const float* c21_w = (const float*)d_in[45];  const float* c21_b = (const float*)d_in[46];
  const float* bn21_g = (const float*)d_in[47]; const float* bn21_b = (const float*)d_in[48];
  const float* c22_w = (const float*)d_in[49];  const float* c22_b = (const float*)d_in[50];
  const float* projD_w = (const float*)d_in[51];
  const float* projD_b = (const float*)d_in[52];
  const float* c0D_w = (const float*)d_in[53];  const float* c0D_b = (const float*)d_in[54];
  const float* bn0D_g = (const float*)d_in[55]; const float* bn0D_b = (const float*)d_in[56];
  const float* c11D_w = (const float*)d_in[57]; const float* c11D_b = (const float*)d_in[58];
  const float* bn11D_g = (const float*)d_in[59]; const float* bn11D_b = (const float*)d_in[60];
  const float* c12D_w = (const float*)d_in[61]; const float* c12D_b = (const float*)d_in[62];
  const float* bn12D_g = (const float*)d_in[63]; const float* bn12D_b = (const float*)d_in[64];
  const float* c21D_w = (const float*)d_in[65]; const float* c21D_b = (const float*)d_in[66];
  const float* bn21D_g = (const float*)d_in[67]; const float* bn21D_b = (const float*)d_in[68];
  const float* c22D_w = (const float*)d_in[69]; const float* c22D_b = (const float*)d_in[70];
  const float* bn22D_g = (const float*)d_in[71]; const float* bn22D_b = (const float*)d_in[72];
  const float* c3c_w = (const float*)d_in[73];  const float* c3c_b = (const float*)d_in[74];
  const float* c3r_w = (const float*)d_in[75];  const float* c3r_b = (const float*)d_in[76];

  // ---- workspace arena ----
  char* wsb = (char*)d_ws;
  size_t off = 0;
  auto alloc = [&](size_t bytes) -> void* {
    void* p = wsb + off;
    off = (off + bytes + 255) & ~(size_t)255;
    return p;
  };
  const size_t T = 131072;
  _Float16* pBfin = (_Float16*)alloc(8 * 512 * 2);          // in_proj  N=128 K=32
  _Float16* pBfxp = (_Float16*)alloc(5 * 2 * 512 * 2);      // x_proj   N=80  K=64
  _Float16* pBfop = (_Float16*)alloc(2 * 2 * 512 * 2);      // out_proj N=32  K=64
  _Float16* pBfbk = (_Float16*)alloc(2 * 512 * 2);          // blk      N=32  K=32
  _Float16* pBsin = (_Float16*)alloc(8 * 512 * 2);
  _Float16* pBsxp = (_Float16*)alloc(5 * 2 * 512 * 2);
  _Float16* pBsop = (_Float16*)alloc(2 * 2 * 512 * 2);
  _Float16* pBsbk = (_Float16*)alloc(2 * 512 * 2);
  _Float16* xt16 = (_Float16*)alloc(T * 32 * 2);
  _Float16* xz16 = (_Float16*)alloc(T * 128 * 2);
  _Float16* xc16 = (_Float16*)alloc(T * 64 * 2);
  float*    xdbl = (float*)alloc(T * 80 * 4);
  _Float16* yg16 = (_Float16*)alloc(T * 64 * 2);
  _Float16* t116 = (_Float16*)alloc(T * 32 * 2);
  _Float16* oh16 = (_Float16*)alloc(T * 32 * 2);
  float* chirp = (float*)alloc(256 * 32 * 4);
  float* xenc  = (float*)alloc(256 * 32 * 4);
  float* feat  = (float*)alloc(2048 * 4);
  float* cb0   = (float*)alloc(131072 * 4);
  float* cb1   = (float*)alloc(131072 * 4);

  auto pack = [&](const float* W, _Float16* out, int Nsrc, int Npad, int K) {
    int total = (Npad / 16) * (K / 32) * 512;
    k_packB<<<(total + 255) / 256, 256, 0, stream>>>(W, out, Nsrc, K, K / 32, total);
  };
  pack(f_inw, pBfin, 128, 128, 32);
  pack(f_xpw, pBfxp, 66, 80, 64);
  pack(f_opw, pBfop, 32, 32, 64);
  pack(f_bkw, pBfbk, 32, 32, 32);
  pack(s_inw, pBsin, 128, 128, 32);
  pack(s_xpw, pBsxp, 66, 80, 64);
  pack(s_opw, pBsop, 32, 32, 64);
  pack(s_bkw, pBsbk, 32, 32, 32);

  auto gemm = [&](const _Float16* A, const _Float16* Bp, float* Cf, _Float16* Ch,
                  const float* bias, int Tn, int K, int N, int act) {
    dim3 g(Tn / 128, N / 16);
    k_gemm<<<g, 256, 0, stream>>>(A, Bp, Cf, Ch, bias, Tn, K, N, act);
  };

  auto run_mamba = [&](int Tn, int L, const _Float16* xt,
                       const _Float16* pin, const _Float16* pxp,
                       const _Float16* pop, const _Float16* pbk,
                       const float* cw, const float* cbias,
                       const float* dtw, const float* dtb,
                       const float* Alog, const float* Dp, const float* bkb,
                       _Float16* outh, float* outf) {
    gemm(xt, pin, nullptr, xz16, nullptr, Tn, 32, 128, 0);                    // in_proj
    k_conv1d<<<(Tn * 64 + 255) / 256, 256, 0, stream>>>(xz16, xc16, cw, cbias, Tn, L);
    gemm(xc16, pxp, xdbl, nullptr, nullptr, Tn, 64, 80, 0);                   // x_proj
    // fused dt + scan + gate
    k_scan<<<Tn / L, 64, 0, stream>>>(xdbl, xc16, xz16, dtw, dtb, Alog, Dp, yg16, L);
    gemm(yg16, pop, nullptr, t116, nullptr, Tn, 64, 32, 0);                   // out_proj
    gemm(t116, pbk, outf, outh, bkb, Tn, 32, 32, 1);                          // blk + silu
  };

  // ---- fast path ----
  k_fast_tokens<<<T / 256, 256, 0, stream>>>(X, in_g, in_b, fast_lin_w, fast_lin_b,
                                             fast_ln_g, fast_ln_b, fast_pe, xt16);
  run_mamba((int)T, 512, xt16, pBfin, pBfxp, pBfop, pBfbk,
            f_cw, f_cb, f_dtw, f_dtb, f_Alog, f_D, f_bkb, oh16, nullptr);
  k_chirp<<<(256 * 32 + 255) / 256, 256, 0, stream>>>(oh16, chirp);

  // ---- slow path ----
  k_slowtok<<<(8192 + 255) / 256, 256, 0, stream>>>(chirp, slow_pe, xt16);
  run_mamba(256, 256, xt16, pBsin, pBsxp, pBsop, pBsbk,
            s_cw, s_cb, s_dtw, s_dtb, s_Alog, s_D, s_bkb, oh16, xenc);

  // ---- feature projections ----
  k_feat<<<1, 256, 0, stream>>>(xenc, chirp, proj_w, proj_b, projD_w, projD_b, feat);
  float* featD = feat + 1024;

  float* seg = (float*)d_out;                 // (1,1,256,224)
  float* det = (float*)d_out + 256 * 224;     // (1,3,128,224)

  auto conv = [&](const float* in, float* out, const float* w, const float* b,
                  const float* g, const float* bb, int Ci, int Co, int H, int W, int act) {
    int total = Co * H * W;
    k_conv3<<<(total + 255) / 256, 256, 0, stream>>>(in, out, w, b, g, bb, Ci, Co, H, W, act);
  };
  auto up = [&](const float* in, float* out, int C, int Hi, int Wi, int Ho, int Wo) {
    int total = C * Ho * Wo;
    k_up<<<(total + 255) / 256, 256, 0, stream>>>(in, out, C, Hi, Wi, Ho, Wo);
  };

  // ---- decode (seg) ----
  conv(feat, cb0, c0_w, c0_b, bn0_g, bn0_b, 1, 8, 32, 32, 1);
  up(cb0, cb1, 8, 32, 32, 64, 64);
  conv(cb1, cb0, c11_w, c11_b, bn11_g, bn11_b, 8, 8, 64, 64, 1);
  conv(cb0, cb1, c12_w, c12_b, bn12_g, bn12_b, 8, 4, 64, 64, 1);
  up(cb1, cb0, 4, 64, 64, 128, 128);
  conv(cb0, cb1, c21_w, c21_b, bn21_g, bn21_b, 4, 4, 128, 128, 1);
  conv(cb1, cb0, c22_w, c22_b, nullptr, nullptr, 4, 1, 128, 128, 2);
  up(cb0, seg, 1, 128, 128, 256, 224);

  // ---- detect (det) ----
  conv(featD, cb0, c0D_w, c0D_b, bn0D_g, bn0D_b, 1, 8, 32, 32, 1);
  up(cb0, cb1, 8, 32, 32, 48, 64);
  conv(cb1, cb0, c11D_w, c11D_b, bn11D_g, bn11D_b, 8, 8, 48, 64, 1);
  conv(cb0, cb1, c12D_w, c12D_b, bn12D_g, bn12D_b, 8, 4, 48, 64, 1);
  up(cb1, cb0, 4, 48, 64, 64, 112);
  conv(cb0, cb1, c21D_w, c21D_b, bn21D_g, bn21D_b, 4, 4, 64, 112, 1);
  conv(cb1, cb0, c22D_w, c22D_b, bn22D_g, bn22D_b, 4, 3, 64, 112, 1);
  up(cb0, cb1, 3, 64, 112, 128, 224);
  conv(cb1, det, c3c_w, c3c_b, nullptr, nullptr, 3, 1, 128, 224, 3);
  conv(cb1, det + 128 * 224, c3r_w, c3r_b, nullptr, nullptr, 3, 2, 128, 224, 0);
}
